// Encoder_83614423318750
// MI455X (gfx1250) — compile-verified, matrix operand order checked by reading, not run on hardware
//
#include <hip/hip_runtime.h>

#define N_NODES 50000
#define IN_C    2000
#define CH      64
#define NH      4
#define NE      800000
#define HC      256
#define ETOT    (NE + N_NODES)

typedef __attribute__((ext_vector_type(16))) __bf16          v16bf;
typedef __attribute__((ext_vector_type(8)))  float           v8f;
typedef __attribute__((ext_vector_type(16))) short           short16;
typedef __attribute__((ext_vector_type(8)))  unsigned short  ushort8v;

__device__ __forceinline__ short f2bf(float f) {
  return (short)(__float_as_uint(f) >> 16);
}

// ---- branchless fragment loaders (A: 16x32 bf16 from f32; B: 32x16 bf16) ----
// lane: m/n = l&15, kb = (l>>4)*8 ; element j<8 -> k=ka+j ; j>=8 -> k=ka+16+(j-8)

__device__ __forceinline__ v16bf a_frag_fast(const float* __restrict__ row, int ka) {
  short16 s;
  float4 f0 = reinterpret_cast<const float4*>(row + ka)[0];
  float4 f1 = reinterpret_cast<const float4*>(row + ka + 4)[0];
  float4 f2 = reinterpret_cast<const float4*>(row + ka + 16)[0];
  float4 f3 = reinterpret_cast<const float4*>(row + ka + 20)[0];
  s[0]=f2bf(f0.x);  s[1]=f2bf(f0.y);  s[2]=f2bf(f0.z);  s[3]=f2bf(f0.w);
  s[4]=f2bf(f1.x);  s[5]=f2bf(f1.y);  s[6]=f2bf(f1.z);  s[7]=f2bf(f1.w);
  s[8]=f2bf(f2.x);  s[9]=f2bf(f2.y);  s[10]=f2bf(f2.z); s[11]=f2bf(f2.w);
  s[12]=f2bf(f3.x); s[13]=f2bf(f3.y); s[14]=f2bf(f3.z); s[15]=f2bf(f3.w);
  return __builtin_bit_cast(v16bf, s);
}

// AE tail (k0 = 1984, K = 2000): group0 (ka..ka+7) always in bounds,
// group1 (ka+16..) always out -> zero. No comparisons, no divergence.
__device__ __forceinline__ v16bf a_frag_ae_tail(const float* __restrict__ row, int ka) {
  short16 s;
  float4 f0 = reinterpret_cast<const float4*>(row + ka)[0];
  float4 f1 = reinterpret_cast<const float4*>(row + ka + 4)[0];
  s[0]=f2bf(f0.x); s[1]=f2bf(f0.y); s[2]=f2bf(f0.z); s[3]=f2bf(f0.w);
  s[4]=f2bf(f1.x); s[5]=f2bf(f1.y); s[6]=f2bf(f1.z); s[7]=f2bf(f1.w);
#pragma unroll
  for (int j = 8; j < 16; ++j) s[j] = (short)0;
  return __builtin_bit_cast(v16bf, s);
}

__device__ __forceinline__ v16bf a_frag_bias(const float* __restrict__ row,
                                             const float* __restrict__ bias, int ka) {
  short16 s;
  float4 f0 = reinterpret_cast<const float4*>(row + ka)[0];
  float4 f1 = reinterpret_cast<const float4*>(row + ka + 4)[0];
  float4 f2 = reinterpret_cast<const float4*>(row + ka + 16)[0];
  float4 f3 = reinterpret_cast<const float4*>(row + ka + 20)[0];
  float4 b0 = reinterpret_cast<const float4*>(bias + ka)[0];
  float4 b1 = reinterpret_cast<const float4*>(bias + ka + 4)[0];
  float4 b2 = reinterpret_cast<const float4*>(bias + ka + 16)[0];
  float4 b3 = reinterpret_cast<const float4*>(bias + ka + 20)[0];
  s[0]=f2bf(f0.x+b0.x);  s[1]=f2bf(f0.y+b0.y);  s[2]=f2bf(f0.z+b0.z);  s[3]=f2bf(f0.w+b0.w);
  s[4]=f2bf(f1.x+b1.x);  s[5]=f2bf(f1.y+b1.y);  s[6]=f2bf(f1.z+b1.z);  s[7]=f2bf(f1.w+b1.w);
  s[8]=f2bf(f2.x+b2.x);  s[9]=f2bf(f2.y+b2.y);  s[10]=f2bf(f2.z+b2.z); s[11]=f2bf(f2.w+b2.w);
  s[12]=f2bf(f3.x+b3.x); s[13]=f2bf(f3.y+b3.y); s[14]=f2bf(f3.z+b3.z); s[15]=f2bf(f3.w+b3.w);
  return __builtin_bit_cast(v16bf, s);
}

__device__ __forceinline__ v16bf b_frag_fast(const unsigned short* __restrict__ row, int ka) {
  short16 s;
  ushort8v u0 = *reinterpret_cast<const ushort8v*>(row + ka);
  ushort8v u1 = *reinterpret_cast<const ushort8v*>(row + ka + 16);
#pragma unroll
  for (int j = 0; j < 8; ++j) { s[j] = (short)u0[j]; s[8 + j] = (short)u1[j]; }
  return __builtin_bit_cast(v16bf, s);
}

__device__ __forceinline__ v16bf b_frag_ae_tail(const unsigned short* __restrict__ row, int ka) {
  short16 s;
  ushort8v u0 = *reinterpret_cast<const ushort8v*>(row + ka);
#pragma unroll
  for (int j = 0; j < 8; ++j) { s[j] = (short)u0[j]; s[8 + j] = (short)0; }
  return __builtin_bit_cast(v16bf, s);
}

// ---- weight conversion to bf16 (one-shot, tiny) ----
__global__ __launch_bounds__(256) void k_convert(
    const float* __restrict__ aew, const float* __restrict__ wl,
    const float* __restrict__ wr, const float* __restrict__ linw,
    unsigned short* __restrict__ aew_bf, unsigned short* __restrict__ wcat_bf,
    unsigned short* __restrict__ linw_bf) {
  int i = blockIdx.x * 256 + threadIdx.x;
  if (i < CH * IN_C) aew_bf[i] = (unsigned short)f2bf(aew[i]);
  int j = i - CH * IN_C;
  if (j >= 0 && j < HC * CH) wcat_bf[j] = (unsigned short)f2bf(wl[j]);
  int j2 = j - HC * CH;
  if (j2 >= 0 && j2 < HC * CH) wcat_bf[HC * CH + j2] = (unsigned short)f2bf(wr[j2]);
  int j3 = j2 - HC * CH;
  if (j3 >= 0 && j3 < CH * HC) linw_bf[j3] = (unsigned short)f2bf(linw[j3]);
}

// ---- h = x @ ae_w.T + ae_b : one wave = 16 rows x 64 cols (4 tiles, shared A) ----
__global__ __launch_bounds__(256) void k_ae_gemm(
    const float* __restrict__ x, const unsigned short* __restrict__ wbf,
    const float* __restrict__ ae_b, float* __restrict__ hout) {
  int wid = (blockIdx.x * 256 + threadIdx.x) >> 5;
  int lane = threadIdx.x & 31;
  if (wid >= 3125) return;
  int tm = wid;
  int kb = (lane >> 4) << 3;
  const float* arow = x + (size_t)(tm * 16 + (lane & 15)) * IN_C;
  const unsigned short* b0 = wbf + (size_t)(0 * 16 + (lane & 15)) * IN_C;
  const unsigned short* b1 = wbf + (size_t)(1 * 16 + (lane & 15)) * IN_C;
  const unsigned short* b2 = wbf + (size_t)(2 * 16 + (lane & 15)) * IN_C;
  const unsigned short* b3 = wbf + (size_t)(3 * 16 + (lane & 15)) * IN_C;
  v8f acc0 = {}, acc1 = {}, acc2 = {}, acc3 = {};
  for (int k0 = 0; k0 < 1984; k0 += 32) {
    __builtin_prefetch(arow + k0 + 256, 0, 0);   // stream x ahead into GL2
    int ka = k0 + kb;
    // issue ALL loads first so the backend can clause them and drain
    // loadcnt progressively across the 4 WMMAs
    v16bf a  = a_frag_fast(arow, ka);
    v16bf f0 = b_frag_fast(b0, ka);
    v16bf f1 = b_frag_fast(b1, ka);
    v16bf f2 = b_frag_fast(b2, ka);
    v16bf f3 = b_frag_fast(b3, ka);
    acc0 = __builtin_amdgcn_wmma_f32_16x16x32_bf16(false, a, false, f0, (short)0, acc0, false, false);
    acc1 = __builtin_amdgcn_wmma_f32_16x16x32_bf16(false, a, false, f1, (short)0, acc1, false, false);
    acc2 = __builtin_amdgcn_wmma_f32_16x16x32_bf16(false, a, false, f2, (short)0, acc2, false, false);
    acc3 = __builtin_amdgcn_wmma_f32_16x16x32_bf16(false, a, false, f3, (short)0, acc3, false, false);
  }
  {  // tail: k0 = 1984, valid K remainder = 16 (branchless, zero-padded)
    int ka = 1984 + kb;
    v16bf a  = a_frag_ae_tail(arow, ka);
    v16bf f0 = b_frag_ae_tail(b0, ka);
    v16bf f1 = b_frag_ae_tail(b1, ka);
    v16bf f2 = b_frag_ae_tail(b2, ka);
    v16bf f3 = b_frag_ae_tail(b3, ka);
    acc0 = __builtin_amdgcn_wmma_f32_16x16x32_bf16(false, a, false, f0, (short)0, acc0, false, false);
    acc1 = __builtin_amdgcn_wmma_f32_16x16x32_bf16(false, a, false, f1, (short)0, acc1, false, false);
    acc2 = __builtin_amdgcn_wmma_f32_16x16x32_bf16(false, a, false, f2, (short)0, acc2, false, false);
    acc3 = __builtin_amdgcn_wmma_f32_16x16x32_bf16(false, a, false, f3, (short)0, acc3, false, false);
  }
  int rm = tm * 16 + ((lane >> 4) << 3);
  int cn = lane & 15;
  v8f accs[4] = {acc0, acc1, acc2, acc3};
#pragma unroll
  for (int t = 0; t < 4; ++t) {
    float bias = ae_b[t * 16 + cn];
#pragma unroll
    for (int r = 0; r < 8; ++r)
      hout[(size_t)(rm + r) * CH + t * 16 + cn] = accs[t][r] + bias;
  }
}

// ---- xl|xr = h @ [wl;wr].T -> [N, 512] : wave = 16 rows x 64 cols ----
__global__ __launch_bounds__(256) void k_xlxr_gemm(
    const float* __restrict__ h, const unsigned short* __restrict__ wcat,
    float* __restrict__ xlxr) {
  int wid = (blockIdx.x * 256 + threadIdx.x) >> 5;
  int lane = threadIdx.x & 31;
  if (wid >= 3125 * 8) return;
  int tm = wid >> 3, g = wid & 7;        // g: group of 4 col-tiles (64 cols)
  int kb = (lane >> 4) << 3;
  const float* arow = h + (size_t)(tm * 16 + (lane & 15)) * CH;
  const unsigned short* b0 = wcat + (size_t)((g * 4 + 0) * 16 + (lane & 15)) * CH;
  const unsigned short* b1 = wcat + (size_t)((g * 4 + 1) * 16 + (lane & 15)) * CH;
  const unsigned short* b2 = wcat + (size_t)((g * 4 + 2) * 16 + (lane & 15)) * CH;
  const unsigned short* b3 = wcat + (size_t)((g * 4 + 3) * 16 + (lane & 15)) * CH;
  v8f acc0 = {}, acc1 = {}, acc2 = {}, acc3 = {};
#pragma unroll
  for (int k0 = 0; k0 < CH; k0 += 32) {
    int ka = k0 + kb;
    v16bf a  = a_frag_fast(arow, ka);
    v16bf f0 = b_frag_fast(b0, ka);
    v16bf f1 = b_frag_fast(b1, ka);
    v16bf f2 = b_frag_fast(b2, ka);
    v16bf f3 = b_frag_fast(b3, ka);
    acc0 = __builtin_amdgcn_wmma_f32_16x16x32_bf16(false, a, false, f0, (short)0, acc0, false, false);
    acc1 = __builtin_amdgcn_wmma_f32_16x16x32_bf16(false, a, false, f1, (short)0, acc1, false, false);
    acc2 = __builtin_amdgcn_wmma_f32_16x16x32_bf16(false, a, false, f2, (short)0, acc2, false, false);
    acc3 = __builtin_amdgcn_wmma_f32_16x16x32_bf16(false, a, false, f3, (short)0, acc3, false, false);
  }
  int rm = tm * 16 + ((lane >> 4) << 3);
  int cn = lane & 15;
  v8f accs[4] = {acc0, acc1, acc2, acc3};
#pragma unroll
  for (int t = 0; t < 4; ++t)
#pragma unroll
    for (int r = 0; r < 8; ++r)
      xlxr[(size_t)(rm + r) * 512 + (g * 4 + t) * 16 + cn] = accs[t][r];
}

// ---- init smax(-inf)/ssum(0)/agg(0) ----
__global__ __launch_bounds__(256) void k_init(unsigned int* __restrict__ smax,
                                              float* __restrict__ ssum,
                                              float* __restrict__ agg) {
  int i = blockIdx.x * 256 + threadIdx.x;
  if (i < N_NODES * NH) { smax[i] = 0xFF800000u; ssum[i] = 0.f; }
  int j = i - N_NODES * NH;
  if (j >= 0 && j < N_NODES * HC) agg[j] = 0.f;
}

// ---- per-edge attention logits + scatter-max (wave per edge) ----
__global__ __launch_bounds__(256) void k_score(
    const long long* __restrict__ ei, const float* __restrict__ xlxr,
    const float* __restrict__ att, float* __restrict__ score, float* __restrict__ smax) {
  int e = (blockIdx.x * 256 + threadIdx.x) >> 5;
  int lane = threadIdx.x & 31;
  if (e >= ETOT) return;
  int src, dst;
  if (e < NE) { src = (int)ei[e]; dst = (int)ei[NE + e]; }
  else        { src = e - NE;     dst = src; }
  int hh = lane >> 3;
  int c0 = (lane & 7) << 3;
  const float4* pl = reinterpret_cast<const float4*>(xlxr + (size_t)src * 512 + hh * 64 + c0);
  const float4* pr = reinterpret_cast<const float4*>(xlxr + (size_t)dst * 512 + 256 + hh * 64 + c0);
  const float4* pa = reinterpret_cast<const float4*>(att + hh * 64 + c0);
  float s = 0.f;
#pragma unroll
  for (int q = 0; q < 2; ++q) {
    float4 a = pl[q], b = pr[q], w = pa[q];
    float t;
    t = a.x + b.x; t = (t >= 0.f) ? t : 0.2f * t; s = fmaf(w.x, t, s);
    t = a.y + b.y; t = (t >= 0.f) ? t : 0.2f * t; s = fmaf(w.y, t, s);
    t = a.z + b.z; t = (t >= 0.f) ? t : 0.2f * t; s = fmaf(w.z, t, s);
    t = a.w + b.w; t = (t >= 0.f) ? t : 0.2f * t; s = fmaf(w.w, t, s);
  }
  s += __shfl_xor(s, 1);
  s += __shfl_xor(s, 2);
  s += __shfl_xor(s, 4);
  if ((lane & 7) == 0) {
    score[(size_t)e * NH + hh] = s;
    float* sm = smax + (size_t)dst * NH + hh;
    if (s >= 0.f) atomicMax((int*)sm, __float_as_int(s));
    else          atomicMin((unsigned int*)sm, __float_as_uint(s));
  }
}

// ---- ex = exp(score - smax[dst]); ssum += ex ----
__global__ __launch_bounds__(256) void k_expsum(
    const long long* __restrict__ ei, float* __restrict__ score,
    const float* __restrict__ smax, float* __restrict__ ssum) {
  int i = blockIdx.x * 256 + threadIdx.x;
  if (i >= ETOT * NH) return;
  int e = i >> 2, hh = i & 3;
  int dst = (e < NE) ? (int)ei[NE + e] : (e - NE);
  float ex = __expf(score[i] - smax[dst * NH + hh]);
  score[i] = ex;
  atomicAdd(&ssum[dst * NH + hh], ex);
}

// ---- alpha = ex/ssum ; agg[dst] += alpha * xl[src]  (wave per edge) ----
__global__ __launch_bounds__(256) void k_agg(
    const long long* __restrict__ ei, const float* __restrict__ xlxr,
    const float* __restrict__ score, const float* __restrict__ ssum,
    float* __restrict__ agg) {
  int e = (blockIdx.x * 256 + threadIdx.x) >> 5;
  int lane = threadIdx.x & 31;
  if (e >= ETOT) return;
  int src, dst;
  if (e < NE) { src = (int)ei[e]; dst = (int)ei[NE + e]; }
  else        { src = e - NE;     dst = src; }
  int hh = lane >> 3;
  int c0 = (lane & 7) << 3;
  float alpha = score[(size_t)e * NH + hh] / (ssum[(size_t)dst * NH + hh] + 1e-16f);
  const float4* pl = reinterpret_cast<const float4*>(xlxr + (size_t)src * 512 + hh * 64 + c0);
  float* po = agg + (size_t)dst * HC + hh * 64 + c0;
  float4 a0 = pl[0], a1 = pl[1];
  atomicAdd(po + 0, alpha * a0.x);
  atomicAdd(po + 1, alpha * a0.y);
  atomicAdd(po + 2, alpha * a0.z);
  atomicAdd(po + 3, alpha * a0.w);
  atomicAdd(po + 4, alpha * a1.x);
  atomicAdd(po + 5, alpha * a1.y);
  atomicAdd(po + 6, alpha * a1.z);
  atomicAdd(po + 7, alpha * a1.w);
}

// ---- z = relu((agg + gat_b) @ lin_w.T + lin_b) : wave = 16 rows x 64 cols ----
__global__ __launch_bounds__(256) void k_post_gemm(
    const float* __restrict__ agg, const float* __restrict__ gat_b,
    const unsigned short* __restrict__ linw, const float* __restrict__ lin_b,
    float* __restrict__ z) {
  int wid = (blockIdx.x * 256 + threadIdx.x) >> 5;
  int lane = threadIdx.x & 31;
  if (wid >= 3125) return;
  int tm = wid;
  int kb = (lane >> 4) << 3;
  const float* arow = agg + (size_t)(tm * 16 + (lane & 15)) * HC;
  const unsigned short* b0 = linw + (size_t)(0 * 16 + (lane & 15)) * HC;
  const unsigned short* b1 = linw + (size_t)(1 * 16 + (lane & 15)) * HC;
  const unsigned short* b2 = linw + (size_t)(2 * 16 + (lane & 15)) * HC;
  const unsigned short* b3 = linw + (size_t)(3 * 16 + (lane & 15)) * HC;
  v8f acc0 = {}, acc1 = {}, acc2 = {}, acc3 = {};
#pragma unroll
  for (int k0 = 0; k0 < HC; k0 += 32) {
    int ka = k0 + kb;
    v16bf a  = a_frag_bias(arow, gat_b, ka);
    v16bf f0 = b_frag_fast(b0, ka);
    v16bf f1 = b_frag_fast(b1, ka);
    v16bf f2 = b_frag_fast(b2, ka);
    v16bf f3 = b_frag_fast(b3, ka);
    acc0 = __builtin_amdgcn_wmma_f32_16x16x32_bf16(false, a, false, f0, (short)0, acc0, false, false);
    acc1 = __builtin_amdgcn_wmma_f32_16x16x32_bf16(false, a, false, f1, (short)0, acc1, false, false);
    acc2 = __builtin_amdgcn_wmma_f32_16x16x32_bf16(false, a, false, f2, (short)0, acc2, false, false);
    acc3 = __builtin_amdgcn_wmma_f32_16x16x32_bf16(false, a, false, f3, (short)0, acc3, false, false);
  }
  int rm = tm * 16 + ((lane >> 4) << 3);
  int cn = lane & 15;
  v8f accs[4] = {acc0, acc1, acc2, acc3};
#pragma unroll
  for (int t = 0; t < 4; ++t) {
    float bias = lin_b[t * 16 + cn];
#pragma unroll
    for (int r = 0; r < 8; ++r) {
      float v = accs[t][r] + bias;
      z[(size_t)(rm + r) * CH + t * 16 + cn] = (v > 0.f) ? v : 0.f;
    }
  }
}

// ---- out = LayerNorm(z + h)  (wave per node, shuffle reductions) ----
__global__ __launch_bounds__(256) void k_ln(
    const float* __restrict__ z, const float* __restrict__ h,
    const float* __restrict__ lnw, const float* __restrict__ lnb,
    float* __restrict__ out) {
  int i = (blockIdx.x * 256 + threadIdx.x) >> 5;
  int lane = threadIdx.x & 31;
  if (i >= N_NODES) return;
  float r0 = z[(size_t)i * CH + lane]      + h[(size_t)i * CH + lane];
  float r1 = z[(size_t)i * CH + lane + 32] + h[(size_t)i * CH + lane + 32];
  float sum = r0 + r1;
#pragma unroll
  for (int o = 16; o >= 1; o >>= 1) sum += __shfl_xor(sum, o);
  float mean = sum * (1.f / 64.f);
  float d0 = r0 - mean, d1 = r1 - mean;
  float v = d0 * d0 + d1 * d1;
#pragma unroll
  for (int o = 16; o >= 1; o >>= 1) v += __shfl_xor(v, o);
  float rstd = rsqrtf(v * (1.f / 64.f) + 1e-12f);
  out[(size_t)i * CH + lane]      = lnw[lane]      * (d0 * rstd) + lnb[lane];
  out[(size_t)i * CH + lane + 32] = lnw[lane + 32] * (d1 * rstd) + lnb[lane + 32];
}

extern "C" void kernel_launch(void* const* d_in, const int* in_sizes, int n_in,
                              void* d_out, int out_size, void* d_ws, size_t ws_size,
                              hipStream_t stream) {
  const float*     x     = (const float*)d_in[0];
  const long long* ei    = (const long long*)d_in[1];
  const float*     ae_w  = (const float*)d_in[2];
  const float*     ae_b  = (const float*)d_in[3];
  const float*     wl    = (const float*)d_in[4];
  const float*     wr    = (const float*)d_in[5];
  const float*     att   = (const float*)d_in[6];
  const float*     gat_b = (const float*)d_in[7];
  const float*     lin_w = (const float*)d_in[8];
  const float*     lin_b = (const float*)d_in[9];
  const float*     ln_w  = (const float*)d_in[10];
  const float*     ln_b  = (const float*)d_in[11];
  float* out = (float*)d_out;

  char* ws = (char*)d_ws;
  size_t off = 0;
  auto take = [&](size_t bytes) {
    char* p = ws + off;
    off += (bytes + 255) & ~(size_t)255;
    return p;
  };
  unsigned short* aew_bf  = (unsigned short*)take((size_t)CH * IN_C * 2);
  unsigned short* wcat_bf = (unsigned short*)take((size_t)2 * HC * CH * 2);
  unsigned short* linw_bf = (unsigned short*)take((size_t)CH * HC * 2);
  float* hbuf  = (float*)take((size_t)N_NODES * CH * 4);
  float* xlxr  = (float*)take((size_t)N_NODES * 512 * 4);
  float* score = (float*)take((size_t)ETOT * NH * 4);
  float* smax  = (float*)take((size_t)N_NODES * NH * 4);
  float* ssum  = (float*)take((size_t)N_NODES * NH * 4);
  float* agg   = (float*)take((size_t)N_NODES * HC * 4);
  float* zbuf  = (float*)take((size_t)N_NODES * CH * 4);

  int conv_threads = CH * IN_C + 2 * HC * CH + CH * HC;
  k_convert<<<(conv_threads + 255) / 256, 256, 0, stream>>>(ae_w, wl, wr, lin_w,
                                                            aew_bf, wcat_bf, linw_bf);
  k_ae_gemm<<<(3125 * 32 + 255) / 256, 256, 0, stream>>>(x, aew_bf, ae_b, hbuf);
  k_xlxr_gemm<<<(3125 * 8 * 32 + 255) / 256, 256, 0, stream>>>(hbuf, wcat_bf, xlxr);
  int init_threads = N_NODES * NH + N_NODES * HC;
  k_init<<<(init_threads + 255) / 256, 256, 0, stream>>>((unsigned int*)smax, ssum, agg);
  k_score<<<(ETOT * 32 + 255) / 256, 256, 0, stream>>>(ei, xlxr, att, score, smax);
  k_expsum<<<(ETOT * NH + 255) / 256, 256, 0, stream>>>(ei, score, smax, ssum);
  k_agg<<<(ETOT * 32 + 255) / 256, 256, 0, stream>>>(ei, xlxr, score, ssum, agg);
  k_post_gemm<<<(3125 * 32 + 255) / 256, 256, 0, stream>>>(agg, gat_b, linw_bf, lin_b, zbuf);
  k_ln<<<(N_NODES * 32 + 255) / 256, 256, 0, stream>>>(zbuf, hbuf, ln_w, ln_b, out);
}
